// GLM_Encoder_41034117546318
// MI455X (gfx1250) — compile-verified, matrix operand order checked by reading, not run on hardware
//
#include <hip/hip_runtime.h>
#include <hip/hip_bf16.h>

#define T_DATA   100000
#define T_NO     200
#define G_SUB    24
#define E_SYN    2000
#define I_SYN    500
#define COS_B    13
#define OBS_B    29        // OBS_BASIS_NO*2-1
#define OBS_LEN  401       // 2*T_NO+1
#define KPAD_E   2016      // 63 chunks of 32
#define KPAD_I   512       // 16 chunks of 32
#define NTILES   (T_DATA/16)   // 6250
#define TG       (T_DATA*G_SUB)

typedef __attribute__((ext_vector_type(16))) _Float16 v16h;
typedef __attribute__((ext_vector_type(8)))  _Float16 v8h;
typedef __attribute__((ext_vector_type(8)))  float    v8f;

// ---------------------------------------------------------------- prep: basis products
__global__ __launch_bounds__(256)
void prep_kernels(const float* __restrict__ W_syn, const float* __restrict__ W_obs,
                  const float* __restrict__ cosB,  const float* __restrict__ obsB,
                  float* __restrict__ eK, float* __restrict__ iK, float* __restrict__ oK)
{
    int g = blockIdx.x;                      // 0..23
    for (int j = threadIdx.x; j < T_NO; j += blockDim.x) {
        float se = 0.f, si = 0.f;
        for (int b = 0; b < COS_B; ++b) {
            float w0 = W_syn[(g*COS_B + b)*2 + 0];
            float w1 = W_syn[(g*COS_B + b)*2 + 1];
            float cb = cosB[b*T_NO + j];
            se += w0*w0*cb;
            si += w1*w1*cb;
        }
        eK[g*T_NO + j] = se;
        iK[g*T_NO + j] = -si;                // i_kern carries the minus sign
    }
    for (int m = threadIdx.x; m < OBS_LEN; m += blockDim.x) {
        float s = 0.f;
        for (int b = 0; b < OBS_B; ++b) s += W_obs[g*OBS_B + b] * obsB[b*OBS_LEN + m];
        oK[g*OBS_LEN + m] = s;
    }
}

// ---------------------------------------------------------------- prep: f16 B panels (C_syn[1:].T, padded)
__global__ __launch_bounds__(256)
void prep_bt(const float* __restrict__ C_e, const float* __restrict__ C_i,
             _Float16* __restrict__ BTe, _Float16* __restrict__ BTi)
{
    const int NE = 32 * KPAD_E;
    const int NI = 32 * KPAD_I;
    int idx = blockIdx.x * blockDim.x + threadIdx.x;
    if (idx < NE) {
        int n = idx / KPAD_E, k = idx % KPAD_E;
        BTe[idx] = (n < G_SUB && k < E_SYN) ? (_Float16)C_e[(size_t)(n+1)*E_SYN + k] : (_Float16)0.f;
    } else if (idx < NE + NI) {
        int r = idx - NE;
        int n = r / KPAD_I, k = r % KPAD_I;
        BTi[r] = (n < G_SUB && k < I_SYN) ? (_Float16)C_i[(size_t)(n+1)*I_SYN + k] : (_Float16)0.f;
    }
}

// ---------------------------------------------------------------- WMMA GEMM: syn = S @ C[1:].T
// A fragment element mapping (16-bit A 16x32): lane<16 -> K 0..7 & 16..23; lane>=16 -> K 8..15 & 24..31.

// unguarded 8x f32 -> 8x f16
__device__ inline v8h loadA8_fast(const float* __restrict__ row, int c)
{
    const float4 v0 = *(const float4*)(row + c);
    const float4 v1 = *(const float4*)(row + c + 4);
    v8h r;
    r[0]=(_Float16)v0.x; r[1]=(_Float16)v0.y; r[2]=(_Float16)v0.z; r[3]=(_Float16)v0.w;
    r[4]=(_Float16)v1.x; r[5]=(_Float16)v1.y; r[6]=(_Float16)v1.z; r[7]=(_Float16)v1.w;
    return r;
}

// guarded at float4 granularity (Kdim % 4 == 0, c % 8 == 0)
__device__ inline v8h loadA8_guard(const float* __restrict__ row, int c, int Kdim)
{
    v8h r;
    if (c + 8 <= Kdim) {
        r = loadA8_fast(row, c);
    } else if (c + 4 <= Kdim) {
        const float4 v0 = *(const float4*)(row + c);
        r[0]=(_Float16)v0.x; r[1]=(_Float16)v0.y; r[2]=(_Float16)v0.z; r[3]=(_Float16)v0.w;
        for (int i = 4; i < 8; ++i) r[i] = (_Float16)0.f;
    } else {
        for (int i = 0; i < 8; ++i) r[i] = (_Float16)0.f;
    }
    return r;
}

template<int KDIM, int KPAD>
__global__ __launch_bounds__(128)
void gemm_onehot(const float* __restrict__ S, const _Float16* __restrict__ BT,
                 float* __restrict__ out)
{
    constexpr int NFULL = KDIM / 32;         // guard-free chunks
    constexpr bool HAS_TAIL = (KDIM % 32) != 0;

    int warp = threadIdx.x >> 5;
    int lane = threadIdx.x & 31;
    int tile = blockIdx.x * 4 + warp;
    if (tile >= NTILES) return;              // wave-uniform: EXEC stays all-ones for WMMA

    int m0   = tile * 16;
    int nlo  = lane & 15;
    int hi   = lane >> 4;
    const float*    Arow = S  + (size_t)(m0 + nlo) * KDIM + hi*8;
    const _Float16* B0   = BT + (size_t)nlo        * KPAD + hi*8;
    const _Float16* B1   = BT + (size_t)(16 + nlo) * KPAD + hi*8;

    v8f acc0 = {}; v8f acc1 = {};

    #pragma unroll 2
    for (int c = 0; c < NFULL; ++c) {
        int kb = c * 32;
        v8h alo = loadA8_fast(Arow, kb);
        v8h ahi = loadA8_fast(Arow, kb + 16);
        v16h a  = __builtin_shufflevector(alo, ahi, 0,1,2,3,4,5,6,7,8,9,10,11,12,13,14,15);

        v8h b0l = *(const v8h*)(B0 + kb);
        v8h b0h = *(const v8h*)(B0 + kb + 16);
        v8h b1l = *(const v8h*)(B1 + kb);
        v8h b1h = *(const v8h*)(B1 + kb + 16);
        v16h b0 = __builtin_shufflevector(b0l, b0h, 0,1,2,3,4,5,6,7,8,9,10,11,12,13,14,15);
        v16h b1 = __builtin_shufflevector(b1l, b1h, 0,1,2,3,4,5,6,7,8,9,10,11,12,13,14,15);

        acc0 = __builtin_amdgcn_wmma_f32_16x16x32_f16(false, a, false, b0, (short)0, acc0, false, false);
        acc1 = __builtin_amdgcn_wmma_f32_16x16x32_f16(false, a, false, b1, (short)0, acc1, false, false);
    }

    if constexpr (HAS_TAIL) {                // executed once, guards allowed here
        constexpr int kb = NFULL * 32;
        v8h alo = loadA8_guard(Arow - hi*8, kb + hi*8,      KDIM);
        v8h ahi = loadA8_guard(Arow - hi*8, kb + 16 + hi*8, KDIM);
        v16h a  = __builtin_shufflevector(alo, ahi, 0,1,2,3,4,5,6,7,8,9,10,11,12,13,14,15);

        v8h b0l = *(const v8h*)(B0 + kb);
        v8h b0h = *(const v8h*)(B0 + kb + 16);
        v8h b1l = *(const v8h*)(B1 + kb);
        v8h b1h = *(const v8h*)(B1 + kb + 16);
        v16h b0 = __builtin_shufflevector(b0l, b0h, 0,1,2,3,4,5,6,7,8,9,10,11,12,13,14,15);
        v16h b1 = __builtin_shufflevector(b1l, b1h, 0,1,2,3,4,5,6,7,8,9,10,11,12,13,14,15);

        acc0 = __builtin_amdgcn_wmma_f32_16x16x32_f16(false, a, false, b0, (short)0, acc0, false, false);
        acc1 = __builtin_amdgcn_wmma_f32_16x16x32_f16(false, a, false, b1, (short)0, acc1, false, false);
    }

    // D layout: lane = N (0..15), VGPR r -> M = r + 8*hi
    for (int r = 0; r < 8; ++r) {
        int row = m0 + r + 8*hi;
        out[(size_t)row * G_SUB + nlo] = acc0[r];
        if (nlo < G_SUB - 16) out[(size_t)row * G_SUB + 16 + nlo] = acc1[r];
    }
}

// ---------------------------------------------------------------- threefry2x32 (JAX key(42) == {0,42})
__device__ inline unsigned rotl32(unsigned x, int d) { return (x << d) | (x >> (32 - d)); }
__device__ inline void tf_group(unsigned& x0, unsigned& x1, int r0, int r1, int r2, int r3)
{
    x0 += x1; x1 = rotl32(x1, r0); x1 ^= x0;
    x0 += x1; x1 = rotl32(x1, r1); x1 ^= x0;
    x0 += x1; x1 = rotl32(x1, r2); x1 ^= x0;
    x0 += x1; x1 = rotl32(x1, r3); x1 ^= x0;
}
__device__ inline void threefry2x32(unsigned c0, unsigned c1, unsigned& o0, unsigned& o1)
{
    const unsigned k0 = 0u, k1 = 42u, k2 = 0u ^ 42u ^ 0x1BD11BDAu;
    unsigned x0 = c0 + k0, x1 = c1 + k1;
    tf_group(x0, x1, 13, 15, 26, 6);  x0 += k1; x1 += k2 + 1u;
    tf_group(x0, x1, 17, 29, 16, 24); x0 += k2; x1 += k0 + 2u;
    tf_group(x0, x1, 13, 15, 26, 6);  x0 += k0; x1 += k1 + 3u;
    tf_group(x0, x1, 17, 29, 16, 24); x0 += k1; x1 += k2 + 4u;
    tf_group(x0, x1, 13, 15, 26, 6);  x0 += k2; x1 += k0 + 5u;
    o0 = x0; o1 = x1;
}
__device__ inline float jax_uniform(unsigned flat)
{
    const unsigned HALF = TG / 2;            // TG even
    unsigned o0, o1, bits;
    if (flat < HALF) { threefry2x32(flat, flat + HALF, o0, o1); bits = o0; }
    else             { threefry2x32(flat - HALF, flat, o0, o1); bits = o1; }
    float f = __uint_as_float((bits >> 9) | 0x3f800000u) - 1.0f;
    return f < 0.f ? 0.f : f;
}

// ---------------------------------------------------------------- fused convs + sigmoid + bernoulli
__global__ __launch_bounds__(256)
void fuse_conv(const float* __restrict__ Z, const float* __restrict__ syn_e,
               const float* __restrict__ syn_i, const float* __restrict__ eK,
               const float* __restrict__ iK, const float* __restrict__ oK,
               const float* __restrict__ Theta, float* __restrict__ out)
{
    __shared__ float sEK[T_NO], sIK[T_NO], sOK[OBS_LEN];
    __shared__ float sSe[456], sSi[456], sZ[656];

    const int g  = blockIdx.y;
    const int t0 = blockIdx.x * 256;
    const int tid = threadIdx.x;

    for (int i = tid; i < T_NO; i += 256) { sEK[i] = eK[g*T_NO + i]; sIK[i] = iK[g*T_NO + i]; }
    for (int i = tid; i < OBS_LEN; i += 256) sOK[i] = oK[g*OBS_LEN + i];
    for (int i = tid; i < 456; i += 256) {
        int r = t0 - T_NO + i;
        bool ok = (r >= 0) && (r < T_DATA);
        sSe[i] = ok ? syn_e[(size_t)r * G_SUB + g] : 0.f;
        sSi[i] = ok ? syn_i[(size_t)r * G_SUB + g] : 0.f;
    }
    for (int i = tid; i < 656; i += 256) {
        int r = t0 - T_NO + i;
        sZ[i] = (r >= 0 && r < T_DATA) ? Z[r] : 0.f;
    }
    __syncthreads();

    int t = t0 + tid;
    if (t >= T_DATA) return;

    float accf = 0.f;
    for (int j = 0; j < T_NO; ++j)           // syn_e[t-1-j] -> sSe[tid + 199 - j]
        accf += sEK[j] * sSe[tid + 199 - j] + sIK[j] * sSi[tid + 199 - j];

    float acco = 0.f;
    for (int mm = 0; mm < OBS_LEN; ++mm)     // Z[t-m], m = mm-200 -> sZ[tid + 400 - mm]
        acco += sOK[mm] * sZ[tid + 400 - mm];

    float x = accf + acco + Theta[g];
    float P = 1.f / (1.f + expf(-x));

    unsigned flat = (unsigned)(t * G_SUB + g);
    float u = jax_uniform(flat);

    out[flat]      = (u < P) ? 1.f : 0.f;    // Z_out
    out[TG + flat] = P;                       // P
}

// ---------------------------------------------------------------- host launch
extern "C" void kernel_launch(void* const* d_in, const int* in_sizes, int n_in,
                              void* d_out, int out_size, void* d_ws, size_t ws_size,
                              hipStream_t stream)
{
    const float* Z     = (const float*)d_in[0];
    const float* S_e   = (const float*)d_in[1];
    const float* S_i   = (const float*)d_in[2];
    const float* C_e   = (const float*)d_in[3];
    const float* C_i   = (const float*)d_in[4];
    const float* W_syn = (const float*)d_in[5];
    const float* Theta = (const float*)d_in[6];
    const float* W_obs = (const float*)d_in[7];
    const float* cosB  = (const float*)d_in[8];
    const float* obsB  = (const float*)d_in[9];
    (void)in_sizes; (void)n_in; (void)out_size; (void)ws_size;

    char* ws = (char*)d_ws;
    size_t off = 0;
    auto carve = [&](size_t bytes) { void* p = ws + off; off = (off + bytes + 255) & ~(size_t)255; return p; };
    float*    syn_e = (float*)   carve((size_t)T_DATA * G_SUB * 4);
    float*    syn_i = (float*)   carve((size_t)T_DATA * G_SUB * 4);
    float*    eK    = (float*)   carve((size_t)G_SUB * T_NO * 4);
    float*    iK    = (float*)   carve((size_t)G_SUB * T_NO * 4);
    float*    oK    = (float*)   carve((size_t)G_SUB * OBS_LEN * 4);
    _Float16* BTe   = (_Float16*)carve((size_t)32 * KPAD_E * 2);
    _Float16* BTi   = (_Float16*)carve((size_t)32 * KPAD_I * 2);

    prep_kernels<<<G_SUB, 256, 0, stream>>>(W_syn, W_obs, cosB, obsB, eK, iK, oK);

    int nbt = (32*KPAD_E + 32*KPAD_I + 255) / 256;
    prep_bt<<<nbt, 256, 0, stream>>>(C_e, C_i, BTe, BTi);

    int gblocks = (NTILES + 3) / 4;
    gemm_onehot<E_SYN, KPAD_E><<<gblocks, 128, 0, stream>>>(S_e, BTe, syn_e);
    gemm_onehot<I_SYN, KPAD_I><<<gblocks, 128, 0, stream>>>(S_i, BTi, syn_i);

    dim3 cgrid((T_DATA + 255) / 256, G_SUB);
    fuse_conv<<<cgrid, 256, 0, stream>>>(Z, syn_e, syn_i, eK, iK, oK, Theta, (float*)d_out);
}
